// WindowAttention3D_43490838840017
// MI455X (gfx1250) — matrix operand order repaired and checked by reading, not compiled, not run on hardware
//
#include <hip/hip_runtime.h>
#include <hip/hip_bf16.h>

typedef __attribute__((ext_vector_type(16))) _Float16 v16h;
typedef __attribute__((ext_vector_type(8)))  _Float16 v8h_t;
typedef __attribute__((ext_vector_type(8)))  float    v8f;
typedef __attribute__((ext_vector_type(4)))  int      v4i;

#define B_TOT 512
#define SEQ   98
#define SEQP  112           // 7 tiles of 16
#define CH    192
#define NH    6
#define HD    32
#define NW    64
#define ROWS  (B_TOT*SEQ)   // 50176, divisible by 16
#define QKVC  (3*CH)        // 576

#if __has_builtin(__builtin_amdgcn_global_load_async_to_lds_b128) && \
    __has_builtin(__builtin_amdgcn_s_wait_asynccnt)
#define ASYNC_LDS 1
#else
#define ASYNC_LDS 0
#endif

static __device__ __forceinline__ v8f wmma_f16(v16h a, v16h b, v8f c) {
    // emits v_wmma_f32_16x16x32_f16
    return __builtin_amdgcn_wmma_f32_16x16x32_f16(false, a, false, b,
                                                  (short)0, c, false, false);
}

// ---------------------------------------------------------------------------
// Kernel 0: x -> f16, weights -> f16, gather relative-position bias [nh][N][N]
// ---------------------------------------------------------------------------
__global__ void __launch_bounds__(256)
prep_kernel(const float* __restrict__ x, const float* __restrict__ qkv_w,
            const float* __restrict__ out_w, const float* __restrict__ bias_table,
            const int* __restrict__ rel_index, _Float16* __restrict__ x_f16,
            _Float16* __restrict__ w_qkv, _Float16* __restrict__ w_out,
            float* __restrict__ bias)
{
    const int nx = ROWS*CH;          // 9,633,792
    const int n1 = QKVC*CH;          // 110,592
    const int n2 = CH*CH;            // 36,864
    const int n3 = NH*SEQ*SEQ;       // 57,624
    for (int i = blockIdx.x*blockDim.x + threadIdx.x; i < nx+n1+n2+n3;
         i += gridDim.x*blockDim.x) {
        if (i < nx) {
            x_f16[i] = (_Float16)x[i];
        } else if (i < nx+n1) {
            w_qkv[i-nx] = (_Float16)qkv_w[i-nx];
        } else if (i < nx+n1+n2) {
            w_out[i-nx-n1] = (_Float16)out_w[i-nx-n1];
        } else {
            int t  = i - nx - n1 - n2;
            int h  = t / (SEQ*SEQ);
            int ij = t % (SEQ*SEQ);
            bias[t] = bias_table[rel_index[ij]*NH + h];
        }
    }
}

// ---------------------------------------------------------------------------
// Kernel 1: QKV GEMM  [ROWS,CH](f16) x [QKVC,CH]^T(f16) -> f16 [ROWS,QKVC]
// 16x64 output per wave (4 WMMA tiles share one A tile); K=192 in 6 chunks.
// Q slab scaled by d^-0.5.
// ---------------------------------------------------------------------------
__global__ void __launch_bounds__(256)
qkv_gemm(const _Float16* __restrict__ x, const _Float16* __restrict__ w,
         const float* __restrict__ bias, _Float16* __restrict__ qkv)
{
    const int lane = threadIdx.x & 31;
    const int wave = threadIdx.x >> 5;
    const int NB   = QKVC/64;                       // 9 column blocks of 64
    const int job  = blockIdx.x*8 + wave;           // 3136*9 = 28224 jobs exactly
    const int tm   = job / NB, tn = job % NB;
    const int ln = lane & 15, g = lane >> 4;

    const _Float16* xrow = x + (size_t)(tm*16 + ln)*CH;
    const _Float16* wr0  = w + (size_t)(tn*64 +  0 + ln)*CH;
    const _Float16* wr1  = w + (size_t)(tn*64 + 16 + ln)*CH;
    const _Float16* wr2  = w + (size_t)(tn*64 + 32 + ln)*CH;
    const _Float16* wr3  = w + (size_t)(tn*64 + 48 + ln)*CH;
    __builtin_prefetch(wr0, 0, 3);
    __builtin_prefetch(wr2, 0, 3);

    v8f acc0 = {}, acc1 = {}, acc2 = {}, acc3 = {};
    #pragma unroll
    for (int kb = 0; kb < CH; kb += 32) {
        v8h_t c0 = *(const v8h_t*)(xrow + kb + 16*g);
        v8h_t c1 = *(const v8h_t*)(xrow + kb + 16*g + 8);
        v16h a;
        #pragma unroll
        for (int u = 0; u < 8; ++u) { a[u] = c0[u]; a[8+u] = c1[u]; }
        v16h b0 = *(const v16h*)(wr0 + kb + 16*g);
        v16h b1 = *(const v16h*)(wr1 + kb + 16*g);
        v16h b2 = *(const v16h*)(wr2 + kb + 16*g);
        v16h b3 = *(const v16h*)(wr3 + kb + 16*g);
        acc0 = wmma_f16(a, b0, acc0);
        acc1 = wmma_f16(a, b1, acc1);
        acc2 = wmma_f16(a, b2, acc2);
        acc3 = wmma_f16(a, b3, acc3);
    }

    const float scale = 0.17677669529663687f;       // 32^-0.5
    _Float16* orow0 = qkv + (size_t)(tm*16 + 8*g)*QKVC + tn*64 + ln;
    #pragma unroll
    for (int r = 0; r < 8; ++r) {
        _Float16* dst = orow0 + (size_t)r*QKVC;
        int colbase = tn*64 + ln;
        float v0 = acc0[r] + bias[colbase +  0];
        float v1 = acc1[r] + bias[colbase + 16];
        float v2 = acc2[r] + bias[colbase + 32];
        float v3 = acc3[r] + bias[colbase + 48];
        if (colbase +  0 < CH) v0 *= scale;
        if (colbase + 16 < CH) v1 *= scale;
        if (colbase + 32 < CH) v2 *= scale;
        if (colbase + 48 < CH) v3 *= scale;
        dst[ 0] = (_Float16)v0;
        dst[16] = (_Float16)v1;
        dst[32] = (_Float16)v2;
        dst[48] = (_Float16)v3;
    }
}

// ---------------------------------------------------------------------------
// Kernel 2: fused attention per (window b, head h). 8 waves / workgroup.
//   S = qk^T + bias + mask  -> softmax -> O = P V
// LDS: S f32 [112][112] (P f16 overlaid row-by-row) + V^T f16 [32][128]
//      + K panel f16 [112][32] staged via async global->LDS copies.
// ---------------------------------------------------------------------------
__global__ void __launch_bounds__(256)
attn_kernel(const _Float16* __restrict__ qkv, const float* __restrict__ bias,
            const float* __restrict__ mask, _Float16* __restrict__ o)
{
    __shared__ float    S[SEQP*SEQP];               // 50176 bytes
    __shared__ _Float16 Vt[HD*128];                 //  8192 bytes
    __shared__ _Float16 Klds[SEQP*HD];              //  7168 bytes (64KB total)

    const int bh = blockIdx.x;
    const int b  = bh / NH, h = bh % NH;
    const int lane = threadIdx.x & 31, wave = threadIdx.x >> 5;
    const int ln = lane & 15, g = lane >> 4;

    const _Float16* base  = qkv + (size_t)b*SEQ*QKVC;
    const int qoff = h*HD, koff = CH + h*HD, voff = 2*CH + h*HD;
    const float* biash = bias + (size_t)h*SEQ*SEQ;
    const float* maskw = mask + (size_t)(b & (NW-1))*SEQ*SEQ;

    // ---- Stage K panel (98 rows x 32 halves) into LDS: 392 x 16B chunks ----
#if ASYNC_LDS
    for (int c = threadIdx.x; c < SEQ*4; c += 256) {
        int row = c >> 2, part = (c & 3) * 8;
        __builtin_amdgcn_global_load_async_to_lds_b128(
            (__attribute__((address_space(1))) v4i*)(base + (size_t)row*QKVC + koff + part),
            (__attribute__((address_space(3))) v4i*)(Klds + row*HD + part),
            0, 0);
    }
#else
    for (int c = threadIdx.x; c < SEQ*4; c += 256) {
        int row = c >> 2, part = (c & 3) * 8;
        *(v8h_t*)(Klds + row*HD + part) =
            *(const v8h_t*)(base + (size_t)row*QKVC + koff + part);
    }
#endif
    // zero-pad K rows 98..111 (448 halves = 224 dwords)
    for (int idx = threadIdx.x; idx < (SEQP-SEQ)*HD/2; idx += 256)
        ((unsigned*)(Klds + SEQ*HD))[idx] = 0u;

    // ---- Stage V^T into LDS: Vt[n][kk] = V[kk][n], zero-padded kk>=98 ------
    for (int idx = threadIdx.x; idx < HD*128; idx += 256) {
        int kk = idx >> 5;                          // 0..127 (row of V)
        int n  = idx & 31;                          // coalesced over n
        Vt[n*128 + kk] = (kk < SEQ) ? base[(size_t)kk*QKVC + voff + n]
                                    : (_Float16)0.f;
    }

    // ---- Phase 1: S tiles (7x7 tiles of 16x16, K=32 -> one WMMA each) ------
#if ASYNC_LDS
    __builtin_amdgcn_s_wait_asynccnt(0);
#endif
    __syncthreads();

    for (int t = wave; t < 49; t += 8) {
        int ti = t / 7, tj = t % 7;
        v16h a = {};
        int qr = ti*16 + ln;
        if (qr < SEQ) {
            const _Float16* p = base + (size_t)qr*QKVC + qoff;
            v8h_t c0 = *(const v8h_t*)(p + 16*g);
            v8h_t c1 = *(const v8h_t*)(p + 16*g + 8);
            #pragma unroll
            for (int u = 0; u < 8; ++u) { a[u] = c0[u]; a[8+u] = c1[u]; }
        }
        // B tile from LDS K panel (zero-padded): 16 consecutive k per lane
        v16h bm = *(const v16h*)(Klds + (tj*16 + ln)*HD + 16*g);
        v8f c = {};
        c = wmma_f16(a, bm, c);
        #pragma unroll
        for (int r = 0; r < 8; ++r) {
            int i = ti*16 + r + 8*g;
            int j = tj*16 + ln;
            float v = (i < SEQ && j < SEQ)
                    ? c[r] + biash[i*SEQ + j] + maskw[i*SEQ + j]
                    : -1e30f;
            S[i*SEQP + j] = v;
        }
    }
    __syncthreads();

    // ---- Phase 2: row softmax; write P (f16, 128 cols) into own row space --
    {
        const int tid = threadIdx.x;
        if (tid < SEQP) {
            float*    row  = S + tid*SEQP;
            _Float16* prow = (_Float16*)row;        // 128 halves fit in 112 floats
            if (tid < SEQ) {
                float mx = -1e30f;
                for (int j = 0; j < SEQ; ++j) mx = fmaxf(mx, row[j]);
                float s = 0.f;
                for (int j = 0; j < SEQ; ++j) s += __expf(row[j] - mx);
                float inv = 1.0f / s;
                for (int j = 0; j < SEQ; ++j) {     // read row[j] before clobbering
                    float e = __expf(row[j] - mx) * inv;
                    prow[j] = (_Float16)e;
                }
                for (int j = SEQ; j < 128; ++j) prow[j] = (_Float16)0.f;
            } else {
                for (int j = 0; j < 128; ++j) prow[j] = (_Float16)0.f;
            }
        }
    }
    __syncthreads();

    // ---- Phase 3: O = P V  (7x2 tiles, K=128 padded -> 4 WMMAs per tile) ---
    for (int t = wave; t < 14; t += 8) {
        int ti = t >> 1, tj = t & 1;
        v8f acc = {};
        #pragma unroll
        for (int kc = 0; kc < 4; ++kc) {
            int kb = kc*32;
            const _Float16* prow = (const _Float16*)(S + (ti*16 + ln)*SEQP);
            v8h_t c0 = *(const v8h_t*)(prow + kb + 16*g);
            v8h_t c1 = *(const v8h_t*)(prow + kb + 16*g + 8);
            v16h a;
            #pragma unroll
            for (int u = 0; u < 8; ++u) { a[u] = c0[u]; a[8+u] = c1[u]; }
            // B tile from LDS V^T: lane ln = column, halves = 16 consecutive k
            v16h bm = *(const v16h*)(Vt + (tj*16 + ln)*128 + kb + 16*g);
            acc = wmma_f16(a, bm, acc);
        }
        #pragma unroll
        for (int r = 0; r < 8; ++r) {
            int i  = ti*16 + r + 8*g;
            int cc = tj*16 + ln;
            if (i < SEQ)
                o[((size_t)b*SEQ + i)*CH + h*HD + cc] = (_Float16)acc[r];
        }
    }
}

// ---------------------------------------------------------------------------
// Kernel 3: output projection [ROWS,CH] x [CH,CH]^T + bias -> f32 out
// 16x64 output per wave; N=192 -> 3 column blocks.
// ---------------------------------------------------------------------------
__global__ void __launch_bounds__(256)
out_gemm(const _Float16* __restrict__ o, const _Float16* __restrict__ w,
         const float* __restrict__ bias, float* __restrict__ out)
{
    const int lane = threadIdx.x & 31;
    const int wave = threadIdx.x >> 5;
    const int NB   = CH/64;                         // 3 column blocks
    const int job  = blockIdx.x*8 + wave;           // 3136*3 = 9408 jobs exactly
    const int tm   = job / NB, tn = job % NB;
    const int ln = lane & 15, g = lane >> 4;

    const _Float16* orow = o + (size_t)(tm*16 + ln)*CH;
    const _Float16* wr0  = w + (size_t)(tn*64 +  0 + ln)*CH;
    const _Float16* wr1  = w + (size_t)(tn*64 + 16 + ln)*CH;
    const _Float16* wr2  = w + (size_t)(tn*64 + 32 + ln)*CH;
    const _Float16* wr3  = w + (size_t)(tn*64 + 48 + ln)*CH;
    __builtin_prefetch(wr0, 0, 3);
    __builtin_prefetch(wr2, 0, 3);

    v8f acc0 = {}, acc1 = {}, acc2 = {}, acc3 = {};
    #pragma unroll
    for (int kb = 0; kb < CH; kb += 32) {
        v8h_t c0 = *(const v8h_t*)(orow + kb + 16*g);
        v8h_t c1 = *(const v8h_t*)(orow + kb + 16*g + 8);
        v16h a;
        #pragma unroll
        for (int u = 0; u < 8; ++u) { a[u] = c0[u]; a[8+u] = c1[u]; }
        v16h b0 = *(const v16h*)(wr0 + kb + 16*g);
        v16h b1 = *(const v16h*)(wr1 + kb + 16*g);
        v16h b2 = *(const v16h*)(wr2 + kb + 16*g);
        v16h b3 = *(const v16h*)(wr3 + kb + 16*g);
        acc0 = wmma_f16(a, b0, acc0);
        acc1 = wmma_f16(a, b1, acc1);
        acc2 = wmma_f16(a, b2, acc2);
        acc3 = wmma_f16(a, b3, acc3);
    }
    float* dst0 = out + (size_t)(tm*16 + 8*g)*CH + tn*64 + ln;
    const float* bb = bias + tn*64 + ln;
    #pragma unroll
    for (int r = 0; r < 8; ++r) {
        float* dst = dst0 + (size_t)r*CH;
        dst[ 0] = acc0[r] + bb[ 0];
        dst[16] = acc1[r] + bb[16];
        dst[32] = acc2[r] + bb[32];
        dst[48] = acc3[r] + bb[48];
    }
}

// ---------------------------------------------------------------------------
extern "C" void kernel_launch(void* const* d_in, const int* in_sizes, int n_in,
                              void* d_out, int out_size, void* d_ws, size_t ws_size,
                              hipStream_t stream)
{
    const float* x          = (const float*)d_in[0];
    const float* mask       = (const float*)d_in[1];
    const float* qkv_w      = (const float*)d_in[2];
    const float* qkv_b      = (const float*)d_in[3];
    const float* out_w      = (const float*)d_in[4];
    const float* out_b      = (const float*)d_in[5];
    const float* bias_table = (const float*)d_in[6];
    const int*   rel_index  = (const int*)d_in[7];
    float* out = (float*)d_out;

    char* ws = (char*)d_ws;
    _Float16* qkv_f16 = (_Float16*)(ws);                 // 50176*576*2 = 57,802,752
    _Float16* o_f16   = (_Float16*)(ws + 57802752);      // 50176*192*2 = 19,267,584
    _Float16* x_f16   = (_Float16*)(ws + 77070336);      // 50176*192*2 = 19,267,584
    _Float16* w_qkv   = (_Float16*)(ws + 96337920);      // 576*192*2   =    221,184
    _Float16* w_out   = (_Float16*)(ws + 96559104);      // 192*192*2   =     73,728
    float*    biasbuf = (float*)   (ws + 96632832);      // 6*98*98*4   =    230,496

    prep_kernel<<<8192, 256, 0, stream>>>(x, qkv_w, out_w, bias_table, rel_index,
                                          x_f16, w_qkv, w_out, biasbuf);
    // 3136 M-tiles * 9 N-blocks / 8 waves = 3528 blocks
    qkv_gemm<<<3528, 256, 0, stream>>>(x_f16, w_qkv, qkv_b, qkv_f16);
    // one workgroup per (window, head)
    attn_kernel<<<B_TOT*NH, 256, 0, stream>>>(qkv_f16, biasbuf, mask, o_f16);
    // 3136 M-tiles * 3 N-blocks / 8 waves = 1176 blocks
    out_gemm<<<1176, 256, 0, stream>>>(o_f16, w_out, out_b, out);
}
